// AttentionCritic_15857019257028
// MI455X (gfx1250) — matrix operand order repaired and checked by reading, not compiled, not run on hardware
//
#include <hip/hip_runtime.h>
#include <hip/hip_bf16.h>

// ---------------------------------------------------------------------------
// MAAC attention critic, MI455X (gfx1250, wave32).
// bf16 WMMA (16x16x32, f32 accum) for all GEMMs; VALU for the tiny 8x8
// cross-agent attention core; all intermediates bf16 in d_ws (L2-resident
// producer->consumer reuse given 192MB L2).
// ---------------------------------------------------------------------------

#define A_N   8
#define B_N   32768
#define SDIM  128
#define ADIM  32
#define IDIM  160
#define H     128
#define NH    4
#define DH    32

typedef unsigned short bfu;  // bf16 bit pattern
typedef __attribute__((ext_vector_type(8)))  unsigned short v8us;
typedef __attribute__((ext_vector_type(16))) unsigned short v16us;
typedef __attribute__((ext_vector_type(16))) __bf16        v16bf;
typedef __attribute__((ext_vector_type(8)))  float         v8f;

// ---- workspace layout (bf16 elements) -------------------------------------
constexpr size_t OFF_ENCWT  = 0;                                 // [A][H][IDIM]
constexpr size_t SZ_ENCWT   = (size_t)A_N * H * IDIM;            // 163840
constexpr size_t OFF_SENCWT = OFF_ENCWT + SZ_ENCWT;              // [A][H][SDIM]
constexpr size_t SZ_SENCWT  = (size_t)A_N * H * SDIM;            // 131072
constexpr size_t OFF_KEYWT  = OFF_SENCWT + SZ_SENCWT;            // [H][H] (out col = n*DH+d)
constexpr size_t SZ_HEADWT  = (size_t)H * H;                     // 16384
constexpr size_t OFF_SELWT  = OFF_KEYWT + SZ_HEADWT;
constexpr size_t OFF_VALWT  = OFF_SELWT + SZ_HEADWT;
constexpr size_t OFF_C1WT   = OFF_VALWT + SZ_HEADWT;             // [A][H][2H]
constexpr size_t SZ_C1WT    = (size_t)A_N * H * 2 * H;           // 262144
constexpr size_t OFF_C2WT   = OFF_C1WT + SZ_C1WT;                // [A][H][H]
constexpr size_t SZ_C2WT    = (size_t)A_N * H * H;               // 131072
constexpr size_t OFF_S      = OFF_C2WT + SZ_C2WT;                // [A][B][H] activations
constexpr size_t SZ_ACT     = (size_t)A_N * B_N * H;             // 33554432
constexpr size_t OFF_K      = OFF_S   + SZ_ACT;
constexpr size_t OFF_SEL    = OFF_K   + SZ_ACT;
constexpr size_t OFF_V      = OFF_SEL + SZ_ACT;
constexpr size_t OFF_OTH    = OFF_V   + SZ_ACT;
constexpr size_t N_WEIGHT_ELEMS = OFF_S;                         // 737280

// ---- scalar helpers --------------------------------------------------------
__device__ __forceinline__ unsigned short f2bf(float f) {
  union { float f; unsigned int i; } c; c.f = f;
  unsigned int x = c.i;
  x += 0x7fffu + ((x >> 16) & 1u);           // round-to-nearest-even
  return (unsigned short)(x >> 16);
}
__device__ __forceinline__ float bfbits2f(unsigned int u16) {
  union { unsigned int i; float f; } c; c.i = u16 << 16; return c.f;
}
__device__ __forceinline__ float bfx4(const uint4 p, int e) {
  unsigned int w = (e < 2) ? p.x : (e < 4) ? p.y : (e < 6) ? p.z : p.w;
  return bfbits2f((e & 1) ? (w >> 16) : (w & 0xffffu));
}
__device__ __forceinline__ float bfx2(const uint2 p, int e) {
  unsigned int w = (e < 2) ? p.x : p.y;
  return bfbits2f((e & 1) ? (w >> 16) : (w & 0xffffu));
}
__device__ __forceinline__ float lrelu(float x) { return x > 0.f ? x : 0.01f * x; }

// ---- WMMA fragment helpers (wave32 layouts per CDNA5 ISA 7.12.2) -----------
// A (16-bit 16x32): lane<16 holds M=lane, K {kk..kk+7, kk+16..kk+23};
//                   lane>=16 holds M=lane-16, K {kk+8..kk+15, kk+24..kk+31}.
// B (16-bit 32x16): same K split as A per out col.
__device__ __forceinline__ v16bf load_frag(const bfu* p0, const bfu* p1) {
  v8us lo = *(const v8us*)p0;
  v8us hi = *(const v8us*)p1;
  v16us t;
#pragma unroll
  for (int e = 0; e < 8; ++e) { t[e] = lo[e]; t[e + 8] = hi[e]; }
  return __builtin_bit_cast(v16bf, t);
}

__device__ __forceinline__ void zacc(v8f* acc) {
#pragma unroll
  for (int i = 0; i < 8; ++i)
    acc[i] = (v8f){0.f, 0.f, 0.f, 0.f, 0.f, 0.f, 0.f, 0.f};
}

// 64x128 output tile, 4 waves (wave w -> rows w*16..w*16+15, all 128 cols).
// Xs: [64][xstr] activations (bf16, row-major). Ws: [128][wstr] W^T (out,in).
template <int KSTEPS>
__device__ __forceinline__ void gemm64x128(const bfu* Xs, int xstr,
                                           const bfu* Ws, int wstr,
                                           v8f* acc, int tid) {
  const int lane = tid & 31;
  const int wave = tid >> 5;
  const int half = lane >> 4;
  const int l15  = lane & 15;
  const bfu* xrow = Xs + (size_t)(wave * 16 + l15) * xstr;
#pragma unroll
  for (int ks = 0; ks < KSTEPS; ++ks) {
    const int kk = ks * 32;
    v16bf af = load_frag(xrow + kk + half * 8, xrow + kk + 16 + half * 8);
#pragma unroll
    for (int ct = 0; ct < 8; ++ct) {
      const bfu* wrow = Ws + (size_t)(ct * 16 + l15) * wstr + kk + half * 8;
      v16bf bfr = load_frag(wrow, wrow + 16);
      acc[ct] = __builtin_amdgcn_wmma_f32_16x16x32_bf16(
          false, af, false, bfr, (short)0, acc[ct], false, false);
    }
  }
}

// C/D layout: lane&15 -> col, VGPR r -> row (upper half lanes rows +8).
__device__ __forceinline__ void epi_lds(const v8f* acc, const float* bias,
                                        bool do_lrelu, bfu* dst, int dstr, int tid) {
  const int lane = tid & 31, wave = tid >> 5, half = lane >> 4, l15 = lane & 15;
#pragma unroll
  for (int ct = 0; ct < 8; ++ct) {
    const int col = ct * 16 + l15;
    const float bv = bias ? bias[col] : 0.f;
#pragma unroll
    for (int r = 0; r < 8; ++r) {
      const int row = wave * 16 + half * 8 + r;
      float v = acc[ct][r] + bv;
      if (do_lrelu) v = lrelu(v);
      dst[(size_t)row * dstr + col] = f2bf(v);
    }
  }
}

__device__ __forceinline__ void epi_global(const v8f* acc, const float* bias,
                                           bool do_lrelu, bfu* dst,
                                           size_t rowbase, int tid) {
  const int lane = tid & 31, wave = tid >> 5, half = lane >> 4, l15 = lane & 15;
#pragma unroll
  for (int ct = 0; ct < 8; ++ct) {
    const int col = ct * 16 + l15;
    const float bv = bias ? bias[col] : 0.f;
#pragma unroll
    for (int r = 0; r < 8; ++r) {
      const int row = wave * 16 + half * 8 + r;
      float v = acc[ct][r] + bv;
      if (do_lrelu) v = lrelu(v);
      dst[(rowbase + row) * H + col] = f2bf(v);
    }
  }
}

// stage [128][K] bf16 global -> LDS with padded stride, 16B chunks
__device__ __forceinline__ void stage_w128(const bfu* src, int K, bfu* dst,
                                           int dstr, int tid) {
  const int gk = K >> 3;
  for (int t = tid; t < 128 * gk; t += 128) {
    const int row = t / gk;
    const int cg  = t - row * gk;
    *(v8us*)(dst + (size_t)row * dstr + cg * 8) =
        *(const v8us*)(src + (size_t)row * K + cg * 8);
  }
}

// ---------------------------------------------------------------------------
// K0: repack all weights fp32 -> bf16, transposed to [out][in]
// ---------------------------------------------------------------------------
__global__ void prep_weights_kernel(const float* __restrict__ encW,
                                    const float* __restrict__ sencW,
                                    const float* __restrict__ keyW,
                                    const float* __restrict__ selW,
                                    const float* __restrict__ valW,
                                    const float* __restrict__ c1W,
                                    const float* __restrict__ c2W,
                                    bfu* __restrict__ ws) {
  const size_t idx = (size_t)blockIdx.x * blockDim.x + threadIdx.x;
  if (idx >= N_WEIGHT_ELEMS) return;
  if (idx < OFF_SENCWT) {                       // encWT[a][h][i] = encW[a][i][h]
    size_t k = idx;
    int a = (int)(k / (H * IDIM)); int r = (int)(k - (size_t)a * H * IDIM);
    int h = r / IDIM, i = r - h * IDIM;
    ws[OFF_ENCWT + k] = f2bf(encW[((size_t)a * IDIM + i) * H + h]);
  } else if (idx < OFF_KEYWT) {                 // sencWT[a][h][i]
    size_t k = idx - OFF_SENCWT;
    int a = (int)(k / (H * SDIM)); int r = (int)(k - (size_t)a * H * SDIM);
    int h = r / SDIM, i = r - h * SDIM;
    ws[OFF_SENCWT + k] = f2bf(sencW[((size_t)a * SDIM + i) * H + h]);
  } else if (idx < OFF_SELWT) {                 // keyWT[c][h], c = n*DH+d
    size_t k = idx - OFF_KEYWT;
    int c = (int)(k / H), h = (int)(k - (size_t)c * H);
    int n = c / DH, d = c - n * DH;
    ws[OFF_KEYWT + k] = f2bf(keyW[((size_t)n * H + h) * DH + d]);
  } else if (idx < OFF_VALWT) {
    size_t k = idx - OFF_SELWT;
    int c = (int)(k / H), h = (int)(k - (size_t)c * H);
    int n = c / DH, d = c - n * DH;
    ws[OFF_SELWT + k] = f2bf(selW[((size_t)n * H + h) * DH + d]);
  } else if (idx < OFF_C1WT) {
    size_t k = idx - OFF_VALWT;
    int c = (int)(k / H), h = (int)(k - (size_t)c * H);
    int n = c / DH, d = c - n * DH;
    ws[OFF_VALWT + k] = f2bf(valW[((size_t)n * H + h) * DH + d]);
  } else if (idx < OFF_C2WT) {                  // c1WT[a][h][i], i in [0,256)
    size_t k = idx - OFF_C1WT;
    int a = (int)(k / (H * 2 * H)); int r = (int)(k - (size_t)a * H * 2 * H);
    int h = r / (2 * H), i = r - h * 2 * H;
    ws[OFF_C1WT + k] = f2bf(c1W[((size_t)a * 2 * H + i) * H + h]);
  } else {                                      // c2WT[a][h][i]
    size_t k = idx - OFF_C2WT;
    int a = (int)(k / (H * H)); int r = (int)(k - (size_t)a * H * H);
    int h = r / H, i = r - h * H;
    ws[OFF_C2WT + k] = f2bf(c2W[((size_t)a * H + i) * H + h]);
  }
}

// ---------------------------------------------------------------------------
// K1: per (agent, 64-batch tile): sa/s encoders + K/S/V head projections.
// 5 fused WMMA GEMMs; LDS rows padded (+8 bf16) for conflict-free b128 reads.
// ---------------------------------------------------------------------------
__global__ __launch_bounds__(128) void enc_heads_kernel(
    const float* __restrict__ states, const float* __restrict__ actions,
    const float* __restrict__ enc_b, const float* __restrict__ senc_b,
    const float* __restrict__ val_b, const bfu* __restrict__ ws,
    bfu* __restrict__ sOut, bfu* __restrict__ kOut,
    bfu* __restrict__ selOut, bfu* __restrict__ vOut) {
  __shared__ bfu Xs[64 * 168];    // concat(state,action) tile, K=160 pad 168
  __shared__ bfu Ws[128 * 168];   // current weight (transposed), padded
  __shared__ bfu SAs[64 * 136];   // sa tile (bf16), K=128 pad 136
  __shared__ bfu Ss[64 * 136];    // s tile
  const int tid = threadIdx.x;
  const int a = blockIdx.y;
  const size_t rowbase = (size_t)a * B_N + (size_t)blockIdx.x * 64;

  // stage X = [states | actions] as bf16
  for (int t = tid; t < 64 * 40; t += 128) {
    const int row = t / 40, cg = t - row * 40, col = cg * 4;
    const float* src = (col < SDIM)
        ? (states + (rowbase + row) * SDIM + col)
        : (actions + (rowbase + row) * ADIM + (col - SDIM));
    const float4 v = *(const float4*)src;
    bfu* d = Xs + (size_t)row * 168 + col;
    d[0] = f2bf(v.x); d[1] = f2bf(v.y); d[2] = f2bf(v.z); d[3] = f2bf(v.w);
  }
  stage_w128(ws + OFF_ENCWT + (size_t)a * H * IDIM, IDIM, Ws, 168, tid);
  __syncthreads();

  v8f acc[8];
  // GEMM1: sa = lrelu(X @ encW + b)
  zacc(acc);
  gemm64x128<5>(Xs, 168, Ws, 168, acc, tid);
  epi_lds(acc, enc_b + a * H, true, SAs, 136, tid);
  __syncthreads();

  // GEMM2: s = lrelu(states @ sencW + b)   (first 128 cols of Xs)
  stage_w128(ws + OFF_SENCWT + (size_t)a * H * SDIM, SDIM, Ws, 136, tid);
  __syncthreads();
  zacc(acc);
  gemm64x128<4>(Xs, 168, Ws, 136, acc, tid);
  epi_lds(acc, senc_b + a * H, true, Ss, 136, tid);
  epi_global(acc, senc_b + a * H, true, sOut, rowbase, tid);
  __syncthreads();

  // GEMM3: K = sa @ keyW
  stage_w128(ws + OFF_KEYWT, H, Ws, 136, tid);
  __syncthreads();
  zacc(acc);
  gemm64x128<4>(SAs, 136, Ws, 136, acc, tid);
  epi_global(acc, nullptr, false, kOut, rowbase, tid);
  __syncthreads();

  // GEMM4: S = s @ selW
  stage_w128(ws + OFF_SELWT, H, Ws, 136, tid);
  __syncthreads();
  zacc(acc);
  gemm64x128<4>(Ss, 136, Ws, 136, acc, tid);
  epi_global(acc, nullptr, false, selOut, rowbase, tid);
  __syncthreads();

  // GEMM5: V = lrelu(sa @ valW + val_b)
  stage_w128(ws + OFF_VALWT, H, Ws, 136, tid);
  __syncthreads();
  zacc(acc);
  gemm64x128<4>(SAs, 136, Ws, 136, acc, tid);
  epi_global(acc, val_b, true, vOut, rowbase, tid);
}

// ---------------------------------------------------------------------------
// K2: cross-agent attention. thread = (head n, batch b). 8x8 logits (dot-32),
// masked softmax over j!=i, weighted V sum -> other[a][b][n*32+d].
// ---------------------------------------------------------------------------
__global__ __launch_bounds__(256) void attention_kernel(
    const bfu* __restrict__ Kg, const bfu* __restrict__ Sg,
    const bfu* __restrict__ Vg, bfu* __restrict__ Og) {
  const int tid = threadIdx.x;
  const int n = tid >> 6;             // head
  const int bi = tid & 63;
  const size_t b = (size_t)blockIdx.x * 64 + bi;
  const int hb = n * DH;
  const float scale = 0.17677669529663687f;  // 1/sqrt(32)

  float lg[8][8];
#pragma unroll
  for (int i = 0; i < 8; ++i)
#pragma unroll
    for (int j = 0; j < 8; ++j) lg[i][j] = 0.f;

#pragma unroll
  for (int c = 0; c < 4; ++c) {       // d chunks of 8
    uint4 sp[8], kp[8];
#pragma unroll
    for (int i = 0; i < 8; ++i) {
      const size_t off = ((size_t)i * B_N + b) * H + hb + c * 8;
      sp[i] = *(const uint4*)(Sg + off);
      kp[i] = *(const uint4*)(Kg + off);
    }
#pragma unroll
    for (int e = 0; e < 8; ++e) {
      float sv[8], kv[8];
#pragma unroll
      for (int i = 0; i < 8; ++i) { sv[i] = bfx4(sp[i], e); kv[i] = bfx4(kp[i], e); }
#pragma unroll
      for (int i = 0; i < 8; ++i)
#pragma unroll
        for (int j = 0; j < 8; ++j) lg[i][j] = fmaf(sv[i], kv[j], lg[i][j]);
    }
  }

  // masked softmax over j != i (matches -1e9 self-mask in reference)
#pragma unroll
  for (int i = 0; i < 8; ++i) {
    float m = -3.0e38f;
#pragma unroll
    for (int j = 0; j < 8; ++j) if (j != i) m = fmaxf(m, lg[i][j]);
    float s = 0.f;
#pragma unroll
    for (int j = 0; j < 8; ++j) {
      const float e = (j == i) ? 0.f : __expf((lg[i][j] - m) * scale);
      lg[i][j] = e; s += e;
    }
    const float inv = 1.f / s;
#pragma unroll
    for (int j = 0; j < 8; ++j) lg[i][j] *= inv;
  }

#pragma unroll
  for (int c = 0; c < 8; ++c) {       // d chunks of 4
    uint2 vp[8];
#pragma unroll
    for (int j = 0; j < 8; ++j)
      vp[j] = *(const uint2*)(Vg + ((size_t)j * B_N + b) * H + hb + c * 4);
    float o[8][4];
#pragma unroll
    for (int i = 0; i < 8; ++i) { o[i][0] = o[i][1] = o[i][2] = o[i][3] = 0.f; }
#pragma unroll
    for (int j = 0; j < 8; ++j) {
      const float v0 = bfx2(vp[j], 0), v1 = bfx2(vp[j], 1);
      const float v2 = bfx2(vp[j], 2), v3 = bfx2(vp[j], 3);
#pragma unroll
      for (int i = 0; i < 8; ++i) {
        o[i][0] = fmaf(lg[i][j], v0, o[i][0]);
        o[i][1] = fmaf(lg[i][j], v1, o[i][1]);
        o[i][2] = fmaf(lg[i][j], v2, o[i][2]);
        o[i][3] = fmaf(lg[i][j], v3, o[i][3]);
      }
    }
#pragma unroll
    for (int i = 0; i < 8; ++i) {
      uint2 ot;
      ot.x = ((unsigned int)f2bf(o[i][1]) << 16) | f2bf(o[i][0]);
      ot.y = ((unsigned int)f2bf(o[i][3]) << 16) | f2bf(o[i][2]);
      *(uint2*)(Og + ((size_t)i * B_N + b) * H + hb + c * 4) = ot;
    }
  }
}

// ---------------------------------------------------------------------------
// K3: critic MLP per (agent, 64-batch tile): [64x256]W1 -> [64x128]W2 -> dot c3
// ---------------------------------------------------------------------------
__global__ __launch_bounds__(128) void critic_kernel(
    const float* __restrict__ c1_b, const float* __restrict__ c2_b,
    const float* __restrict__ c3_W, const float* __restrict__ c3_b,
    const bfu* __restrict__ ws, const bfu* __restrict__ sIn,
    const bfu* __restrict__ oIn, float* __restrict__ qOut) {
  __shared__ bfu Xc[64 * 264];   // critic_in [64][256] pad 264; reused as h2
  __shared__ bfu Wc[128 * 264];  // weights
  __shared__ bfu Hc[64 * 136];   // h1
  __shared__ float c3s[H];
  const int tid = threadIdx.x;
  const int a = blockIdx.y;
  const size_t rowbase = (size_t)a * B_N + (size_t)blockIdx.x * 64;

  // stage critic_in = [s | other]
  for (int t = tid; t < 64 * 32; t += 128) {
    const int row = t / 32, cg = t - row * 32, col = cg * 8;
    const bfu* src = (col < H) ? (sIn + (rowbase + row) * H + col)
                               : (oIn + (rowbase + row) * H + (col - H));
    *(v8us*)(Xc + (size_t)row * 264 + col) = *(const v8us*)src;
  }
  stage_w128(ws + OFF_C1WT + (size_t)a * H * 2 * H, 2 * H, Wc, 264, tid);
  __syncthreads();

  v8f acc[8];
  zacc(acc);
  gemm64x128<8>(Xc, 264, Wc, 264, acc, tid);
  epi_lds(acc, c1_b + a * H, true, Hc, 136, tid);
  __syncthreads();

  stage_w128(ws + OFF_C2WT + (size_t)a * H * H, H, Wc, 136, tid);
  if (tid < H) c3s[tid] = c3_W[a * H + tid];
  __syncthreads();

  zacc(acc);
  gemm64x128<4>(Hc, 136, Wc, 136, acc, tid);
  epi_lds(acc, c2_b + a * H, true, Xc, 136, tid);   // h2 into Xc
  __syncthreads();

  if (tid < 64) {
    float q = c3_b[a];
#pragma unroll 8
    for (int k = 0; k < H; ++k) q += bfbits2f(Xc[(size_t)tid * 136 + k]) * c3s[k];
    qOut[rowbase + tid] = q;
  }
}

// ---------------------------------------------------------------------------
extern "C" void kernel_launch(void* const* d_in, const int* in_sizes, int n_in,
                              void* d_out, int out_size, void* d_ws, size_t ws_size,
                              hipStream_t stream) {
  (void)in_sizes; (void)n_in; (void)out_size; (void)ws_size;
  const float* states = (const float*)d_in[0];
  const float* actions = (const float*)d_in[1];
  const float* enc_W  = (const float*)d_in[2];
  const float* enc_b  = (const float*)d_in[3];
  const float* senc_W = (const float*)d_in[4];
  const float* senc_b = (const float*)d_in[5];
  const float* key_W  = (const float*)d_in[6];
  const float* sel_W  = (const float*)d_in[7];
  const float* val_W  = (const float*)d_in[8];
  const float* val_b  = (const float*)d_in[9];
  const float* c1_W   = (const float*)d_in[10];
  const float* c1_b   = (const float*)d_in[11];
  const float* c2_W   = (const float*)d_in[12];
  const float* c2_b   = (const float*)d_in[13];
  const float* c3_W   = (const float*)d_in[14];
  const float* c3_b   = (const float*)d_in[15];
  bfu* ws = (bfu*)d_ws;

  prep_weights_kernel<<<(int)((N_WEIGHT_ELEMS + 255) / 256), 256, 0, stream>>>(
      enc_W, senc_W, key_W, sel_W, val_W, c1_W, c2_W, ws);

  dim3 gTiles(B_N / 64, A_N);
  enc_heads_kernel<<<gTiles, 128, 0, stream>>>(
      states, actions, enc_b, senc_b, val_b, ws,
      ws + OFF_S, ws + OFF_K, ws + OFF_SEL, ws + OFF_V);

  attention_kernel<<<B_N / 64, 256, 0, stream>>>(
      ws + OFF_K, ws + OFF_SEL, ws + OFF_V, ws + OFF_OTH);

  critic_kernel<<<gTiles, 128, 0, stream>>>(
      c1_b, c2_b, c3_W, c3_b, ws, ws + OFF_S, ws + OFF_OTH, (float*)d_out);
}
